// HungarianAssigner_16990890623199
// MI455X (gfx1250) — matrix operand order repaired and checked by reading, not compiled, not run on hardware
//
#include <hip/hip_runtime.h>
#include <hip/hip_bf16.h>

typedef __attribute__((ext_vector_type(16))) __bf16 v16bf;
typedef __attribute__((ext_vector_type(8)))  float  v8f;

#define HW_K    65536
#define NB      4
#define NN      100
#define NM      20
#define NC      80
#define NTILES  7          // ceil(100/16)
#define KCHUNKS 32         // K split for occupancy: 4*7*32 = 896 waves
#define KC      (HW_K / KCHUNKS)   // 2048
#define KSTEPS  (KC / 32)          // 64 WMMA K-steps per chunk
#define NPAD    112        // 7*16
#define MPAD    32         // 2*16

__device__ __forceinline__ float sigm(float x) {
    // v_exp_f32 + v_add + v_rcp_f32 (hardware rcp, ~1 ulp; feeds bf16 anyway)
    return __builtin_amdgcn_rcpf(1.0f + __expf(-x));
}

// Raw per-step fragment data: 12 x float4 = 48 VGPRs (double-buffered -> 96).
struct Raw {
    float4 a0, a1, a2, a3;   // A: 16 f32 at kb+16g
    float4 b0, b1, b2, b3;   // B0: 16 f32 at kb+16g (gt rows 0..15)
    float4 c0, c1, c2, c3;   // B1: 16 f32 at kb+16g (gt rows 16..31, zero page)
};

__device__ __forceinline__ Raw load_step(const float* __restrict__ pArow,
                                         const float* __restrict__ gRow0,
                                         const float* __restrict__ gRow1,
                                         int kb, int g) {
    Raw f;
    const float4* a0 = (const float4*)(pArow + kb + 16 * g);
    const float4* a1 = (const float4*)(pArow + kb + 16 * g + 8);
    f.a0 = a0[0]; f.a1 = a0[1]; f.a2 = a1[0]; f.a3 = a1[1];
    const float4* bp = (const float4*)(gRow0 + kb + 16 * g);
    f.b0 = bp[0]; f.b1 = bp[1]; f.b2 = bp[2]; f.b3 = bp[3];
    const float4* cp = (const float4*)(gRow1 + kb + 16 * g);
    f.c0 = cp[0]; f.c1 = cp[1]; f.c2 = cp[2]; f.c3 = cp[3];
    return f;
}

__device__ __forceinline__ void process_step(const Raw& f,
                                             float& ps, float& ts0, float& ts1,
                                             v8f& c0, v8f& c1) {
    // A fragment: sigmoid in f32 (exact row-sum), products in bf16.
    float av[16] = {f.a0.x, f.a0.y, f.a0.z, f.a0.w, f.a1.x, f.a1.y, f.a1.z, f.a1.w,
                    f.a2.x, f.a2.y, f.a2.z, f.a2.w, f.a3.x, f.a3.y, f.a3.z, f.a3.w};
    v16bf a;
#pragma unroll
    for (int j = 0; j < 16; ++j) {
        float sv = sigm(av[j]);
        ps += sv;
        a[j] = (__bf16)sv;
    }

    float bv[16] = {f.b0.x, f.b0.y, f.b0.z, f.b0.w, f.b1.x, f.b1.y, f.b1.z, f.b1.w,
                    f.b2.x, f.b2.y, f.b2.z, f.b2.w, f.b3.x, f.b3.y, f.b3.z, f.b3.w};
    v16bf bb0;
#pragma unroll
    for (int j = 0; j < 16; ++j) {
        ts0 += bv[j];
        bb0[j] = (__bf16)bv[j];
    }

    float cv[16] = {f.c0.x, f.c0.y, f.c0.z, f.c0.w, f.c1.x, f.c1.y, f.c1.z, f.c1.w,
                    f.c2.x, f.c2.y, f.c2.z, f.c2.w, f.c3.x, f.c3.y, f.c3.z, f.c3.w};
    v16bf bb1;
#pragma unroll
    for (int j = 0; j < 16; ++j) {
        ts1 += cv[j];           // invalid rows read a zero page -> adds 0
        bb1[j] = (__bf16)cv[j];
    }

    c0 = __builtin_amdgcn_wmma_f32_16x16x32_bf16(
             false, a, false, bb0, (short)0, c0, false, false);
    c1 = __builtin_amdgcn_wmma_f32_16x16x32_bf16(
             false, a, false, bb1, (short)0, c1, false, false);
}

// Phase A: pt partials via bf16 WMMA (f32 accumulate), plus p_sum / t_sum in f32.
__global__ void __launch_bounds__(256)
hungarian_phaseA(const float* __restrict__ pred,   // (B,N,HW)
                 const float* __restrict__ gt,     // (B,M,HW)
                 float* __restrict__ pt,           // (B,NPAD,MPAD)
                 float* __restrict__ psum,         // (B,NPAD)
                 float* __restrict__ tsum,         // (B,MPAD)
                 const float* __restrict__ zrow) { // KC zeros
    const int tid    = blockIdx.x * blockDim.x + threadIdx.x;
    const int lane   = threadIdx.x & 31;
    const int wid    = tid >> 5;                 // global wave id
    const int kchunk = wid & (KCHUNKS - 1);      // uniform per wave
    const int ntile  = (wid / KCHUNKS) % NTILES; // uniform per wave
    const int b      = wid / (KCHUNKS * NTILES); // uniform per wave
    const int g      = lane >> 4;                // lane half (K select)
    const int r      = lane & 15;                // row/col index within tile
    const int kbase  = kchunk * KC;

    // A: pred row (clamped for padded n; results land in ignored padded slots)
    const int nrow   = ntile * 16 + r;
    const int nrow_c = nrow < NN ? nrow : (NN - 1);
    const float* __restrict__ pArow = pred + (size_t)(b * NN + nrow_c) * HW_K;

    // B0: gt rows 0..15 (always valid, M=20).
    const float* __restrict__ gRow0 = gt + (size_t)(b * NM + r) * HW_K;

    // B1: gt rows 16..19; invalid lanes point at a zero page biased by -kbase,
    // so gRow1 + kb + 16g stays inside zrow[0..KC) for this wave's K-chunk.
    const int  m1row = 16 + r;
    const bool m1v   = (m1row < NM);
    const float* __restrict__ gRow1 =
        m1v ? (gt + (size_t)(b * NM + m1row) * HW_K) : (zrow - kbase);

    v8f c0 = {};   // D tile, m-cols 0..15
    v8f c1 = {};   // D tile, m-cols 16..31
    float ps = 0.0f, ts0 = 0.0f, ts1 = 0.0f;

    // Software-pipelined: issue step s+1 loads before processing step s.
    Raw cur = load_step(pArow, gRow0, gRow1, kbase, g);
    for (int s = 0; s < KSTEPS; ++s) {
        const int nkb = (s + 1 < KSTEPS) ? (kbase + (s + 1) * 32) : kbase;
        Raw nxt = load_step(pArow, gRow0, gRow1, nkb, g);
        process_step(cur, ps, ts0, ts1, c0, c1);
        cur = nxt;
    }

    // Combine lane halves (lanes l and l^16 cover disjoint K for same row/col)
    ps  += __shfl_xor(ps,  16, 32);
    ts0 += __shfl_xor(ts0, 16, 32);
    ts1 += __shfl_xor(ts1, 16, 32);

    // Scatter pt partials. C layout: lane half g, VGPR j -> row j+8g, col r.
    const int pt_base = (b * NPAD + ntile * 16) * MPAD;
#pragma unroll
    for (int j = 0; j < 8; ++j) {
        atomicAdd(&pt[pt_base + (j + 8 * g) * MPAD + r],      c0[j]);
        atomicAdd(&pt[pt_base + (j + 8 * g) * MPAD + 16 + r], c1[j]);
    }
    if (lane < 16) {
        atomicAdd(&psum[b * NPAD + ntile * 16 + r], ps);
        if (ntile == 0) atomicAdd(&tsum[b * MPAD + r], ts0);
    }
    if (lane < 4 && ntile == 0) {
        atomicAdd(&tsum[b * MPAD + 16 + r], ts1);
    }
}

// Phase B: combine pt / p_sum / t_sum with gathered class cost. 8000 outputs.
__global__ void __launch_bounds__(256)
hungarian_phaseB(const float* __restrict__ logits,  // (B,N,C)
                 const int*   __restrict__ labels,  // (B,M)
                 const float* __restrict__ pt,
                 const float* __restrict__ psum,
                 const float* __restrict__ tsum,
                 float* __restrict__ out) {          // (B,N,M)
    const int i = blockIdx.x * blockDim.x + threadIdx.x;
    if (i >= NB * NN * NM) return;
    const int m = i % NM;
    const int n = (i / NM) % NN;
    const int b = i / (NM * NN);

    const float PT = pt[(b * NPAD + n) * MPAD + m];
    const float PS = psum[b * NPAD + n];
    const float TS = tsum[b * MPAD + m];
    const int  lab = labels[b * NM + m];
    const float cc = -logits[(b * NN + n) * NC + lab];

    const float fHW   = (float)HW_K;
    const float cmask = -PT / fHW - (fHW - PS - TS + PT) / fHW;
    const float cdice = 1.0f - (2.0f * PT + 1.0f) / (PS + TS + 1.0f);

    out[i] = cc + cmask + cdice;   // W_CLASS = W_MASK = W_DICE = 1
}

extern "C" void kernel_launch(void* const* d_in, const int* in_sizes, int n_in,
                              void* d_out, int out_size, void* d_ws, size_t ws_size,
                              hipStream_t stream) {
    const float* logits = (const float*)d_in[0];   // pred_logits (B,N,C)
    const float* pred   = (const float*)d_in[1];   // pred_masks  (B,N,H,W)
    const float* gt     = (const float*)d_in[2];   // gt_masks    (B,M,H,W)
    const int*   labels = (const int*)d_in[3];     // gt_labels   (B,M)
    float*       out    = (float*)d_out;

    float* pt   = (float*)d_ws;                    // B*NPAD*MPAD
    float* psum = pt + NB * NPAD * MPAD;           // B*NPAD
    float* tsum = psum + NB * NPAD;                // B*MPAD
    float* zrow = tsum + NB * MPAD;                // KC zeros (B1 zero page)
    const size_t zbytes =
        (size_t)(NB * NPAD * MPAD + NB * NPAD + NB * MPAD + KC) * sizeof(float);

    hipMemsetAsync(d_ws, 0, zbytes, stream);       // graph-capturable

    const int totalThreads = NB * NTILES * KCHUNKS * 32;   // 28672
    hungarian_phaseA<<<totalThreads / 256, 256, 0, stream>>>(pred, gt, pt, psum,
                                                             tsum, zrow);

    const int nOut = NB * NN * NM;                 // 8000
    hungarian_phaseB<<<(nOut + 255) / 256, 256, 0, stream>>>(logits, labels, pt,
                                                             psum, tsum, out);
}
